// MoE_90829968375928
// MI455X (gfx1250) — matrix operand order repaired and checked by reading, not compiled, not run on hardware
//
#include <hip/hip_runtime.h>
#include <math.h>

// ---------------- problem constants (match reference) ----------------
#define TOKENS  (4 * 2048)   // B*S = 8192
#define DMODEL  1024
#define NEXP    8
#define HDIM    4096
#define TOPK    2
#define NROWS   (TOKENS * TOPK)   // 16384 routed rows total

typedef unsigned short bf16r;          // raw bf16 storage
typedef unsigned int   u32;
typedef __attribute__((ext_vector_type(16))) __bf16 v16bf;
typedef __attribute__((ext_vector_type(8)))  float  v8f;

// fp32 -> bf16 round-to-nearest-even
__device__ __forceinline__ bf16r f2bf(float f) {
    u32 u = __float_as_uint(f);
    u += 0x7fffu + ((u >> 16) & 1u);
    return (bf16r)(u >> 16);
}

// ---- CDNA5 async memory->LDS copy (16B per lane), tracked by ASYNCcnt ----
// VDST VGPR holds the wave-relative LDS byte address (= low 32 bits of the
// generic shared-memory pointer, per the LDS aperture rule addr[31:0]).
__device__ __forceinline__ void async_copy_b128(unsigned lds_byte,
                                                const void* gaddr) {
    unsigned long long ga = (unsigned long long)(uintptr_t)gaddr;
    asm volatile("global_load_async_to_lds_b128 %0, %1, off"
                 :: "v"(lds_byte), "v"(ga)
                 : "memory");
}

__device__ __forceinline__ void wait_async0() {
#if __has_builtin(__builtin_amdgcn_s_wait_asynccnt)
    __builtin_amdgcn_s_wait_asynccnt(0);
#else
    asm volatile("s_wait_asynccnt 0x0" ::: "memory");
#endif
}

// ---------------- small setup kernels ----------------
__global__ void zero_counts_kernel(int* counts) {
    if (threadIdx.x < NEXP) counts[threadIdx.x] = 0;
}

__global__ void cvt_bf16_kernel(const float* __restrict__ src,
                                bf16r* __restrict__ dst, int n) {
    for (int i = blockIdx.x * blockDim.x + threadIdx.x; i < n;
         i += gridDim.x * blockDim.x)
        dst[i] = f2bf(src[i]);
}

// Swizzle weights [E][Kdim][Ndim] (row-major, K x N per expert) into the
// WMMA B-fragment layout: [E][Ndim/16][Kdim/32][lane(32)][elem(16)] bf16.
// For element (k,n) inside a 32x16 (KxN) fragment:
//   lane = (n&15) | ((k&8)<<1), elem = (k&7) | ((k&16)>>1)   (elem pairs = packed lo/hi 16-bit)
__global__ void swizzle_w_kernel(const float* __restrict__ src,
                                 bf16r* __restrict__ dst, int Kdim, int Ndim) {
    long long n = (long long)NEXP * Kdim * Ndim;
    for (long long i = blockIdx.x * (long long)blockDim.x + threadIdx.x; i < n;
         i += (long long)gridDim.x * blockDim.x) {
        int nn = (int)(i % Ndim);
        long long r = i / Ndim;
        int k = (int)(r % Kdim);
        int e = (int)(r / Kdim);
        int nt = nn >> 4, kc = k >> 5, kk = k & 31;
        int ln = (nn & 15) | ((kk & 8) << 1);
        int el = (kk & 7) | ((kk & 16) >> 1);
        size_t di = ((((size_t)e * (Ndim >> 4) + nt) * (Kdim >> 5) + kc) << 9)
                    + ((size_t)ln << 4) + el;
        dst[di] = f2bf(src[i]);
    }
}

// ---------------- gating: one wave32 per token ----------------
__global__ __launch_bounds__(256) void gate_topk_kernel(
    const float* __restrict__ x, const float* __restrict__ gw,
    int* __restrict__ tidx, float* __restrict__ tw, int* __restrict__ counts) {
    int wave = threadIdx.x >> 5;
    int lane = threadIdx.x & 31;
    int t = blockIdx.x * 8 + wave;
    float s[NEXP];
#pragma unroll
    for (int e = 0; e < NEXP; ++e) s[e] = 0.f;
    const float* xr = x + (size_t)t * DMODEL;
    for (int j = lane; j < DMODEL; j += 32) {
        float xv = xr[j];
        const float* g = gw + (size_t)j * NEXP;
#pragma unroll
        for (int e = 0; e < NEXP; ++e) s[e] = fmaf(xv, g[e], s[e]);
    }
#pragma unroll
    for (int e = 0; e < NEXP; ++e)
#pragma unroll
        for (int m = 16; m >= 1; m >>= 1) s[e] += __shfl_xor(s[e], m, 32);
    if (lane == 0) {
        int b0 = 0; float v0 = s[0];
#pragma unroll
        for (int e = 1; e < NEXP; ++e) if (s[e] > v0) { v0 = s[e]; b0 = e; }
        int b1 = -1; float v1 = -3.4e38f;
#pragma unroll
        for (int e = 0; e < NEXP; ++e)
            if (e != b0 && s[e] > v1) { v1 = s[e]; b1 = e; }
        float e1 = __expf(v1 - v0);          // softmax over the two picks
        float inv = 1.f / (1.f + e1);
        tidx[t * 2 + 0] = b0; tidx[t * 2 + 1] = b1;
        tw[t * 2 + 0] = inv;  tw[t * 2 + 1] = e1 * inv;
        atomicAdd(&counts[b0], 1);
        atomicAdd(&counts[b1], 1);
    }
}

__global__ void scan_kernel(const int* __restrict__ counts,
                            int* __restrict__ offs, int* __restrict__ cursors) {
    if (threadIdx.x == 0) {
        int acc = 0;
        for (int e = 0; e < NEXP; ++e) {
            offs[e] = acc; cursors[e] = acc; acc += counts[e];
        }
        offs[NEXP] = acc;
    }
}

__global__ void assign_rows_kernel(const int* __restrict__ tidx,
                                   const float* __restrict__ tw,
                                   int* __restrict__ cursors,
                                   int* __restrict__ token_of_row,
                                   float* __restrict__ weight_of_row,
                                   int* __restrict__ row_of) {
    int t = blockIdx.x * blockDim.x + threadIdx.x;
    if (t >= TOKENS) return;
#pragma unroll
    for (int k = 0; k < TOPK; ++k) {
        int e = tidx[t * 2 + k];
        int r = atomicAdd(&cursors[e], 1);
        token_of_row[r] = t;
        weight_of_row[r] = tw[t * 2 + k];
        row_of[t * 2 + k] = r;
    }
}

// ---------------- routed WMMA GEMM ----------------
// Block tile 128x128, K-step 32, 256 threads = 8 waves, wave tile 64x32
// (4x2 fragments of 16x16). A staged into LDS in fragment layout with
// GLOBAL_LOAD_ASYNC_TO_LDS_B128 (double buffered, ASYNCcnt-synchronized);
// B read from pre-swizzled global (weights are L2-resident: 134MB < 192MB).
template <bool FC1>
__global__ __launch_bounds__(256) void moe_gemm_kernel(
    const bf16r* __restrict__ A,       // FC1: xb [T,D]   FC2: h [NROWS,H]
    const bf16r* __restrict__ Bsw,     // swizzled weights
    const float* __restrict__ bias,    // [E, Ndim]
    bf16r* __restrict__ outb,          // FC1 output (bf16, post-GELU)
    float* __restrict__ outf,          // FC2 output (f32)
    const int* __restrict__ offs,      // [E+1] row offsets
    const int* __restrict__ token_of_row,
    int Kdim, int Ndim, int lda) {
    __shared__ alignas(32) bf16r ldsA[2][128 * 32];

    int e = blockIdx.z;
    int r0 = offs[e], r1 = offs[e + 1];
    int mloc = blockIdx.y * 128;
    int nrows = r1 - r0;
    if (mloc >= nrows) return;           // uniform per block: EXEC stays full
    int gm = r0 + mloc;
    int nbase = blockIdx.x * 128;

    int tid = threadIdx.x;
    int lane = tid & 31;
    int wave = tid >> 5;
    int wm = wave >> 2, wn = wave & 3;

    // --- A staging plan for this thread (copies 16 contiguous bf16) ---
    int smt = tid >> 5;                        // m-subtile 0..7
    int sln = tid & 31;
    int si = smt * 16 + (sln & 15);            // local row 0..127
    int sgr = gm + (((mloc + si) < nrows) ? si : 0);   // clamp padding rows
    int arow = FC1 ? token_of_row[sgr] : sgr;  // gather tokens for FC1
    const bf16r* asrc = A + (size_t)arow * lda + ((sln & 16) ? 8 : 0);
    unsigned sdst =
        (unsigned)(uintptr_t)(&ldsA[0][0] + (size_t)(smt * 32 + sln) * 16);

    int KC = Kdim >> 5;
    v8f acc[4][2];
#pragma unroll
    for (int mi = 0; mi < 4; ++mi)
#pragma unroll
        for (int ni = 0; ni < 2; ++ni)
#pragma unroll
            for (int q = 0; q < 8; ++q) acc[mi][ni][q] = 0.f;

    // prologue: async-stage k-chunk 0 into buffer 0
    async_copy_b128(sdst, asrc);
    async_copy_b128(sdst + 16, asrc + 16);
    wait_async0();
    __syncthreads();

    size_t bstride = (size_t)(Kdim >> 5) * 512;   // bf16 per n-tile column
    const bf16r* bbase =
        Bsw + ((size_t)e * (Ndim >> 4) + (nbase >> 4) + wn * 2) * bstride
            + (size_t)lane * 16;

    for (int kc = 0; kc < KC; ++kc) {
        int buf = kc & 1;
        if (kc + 1 < KC) {                 // async-stage next chunk (back buf)
            const bf16r* s = asrc + (size_t)(kc + 1) * 32;
            unsigned d = sdst + (unsigned)(1 - buf) * (128 * 32 * 2);
            async_copy_b128(d, s);
            async_copy_b128(d + 16, s + 16);
            __builtin_prefetch(bbase + (size_t)(kc + 1) * 512, 0, 1);
        }
        v16bf bfrag[2];
#pragma unroll
        for (int ni = 0; ni < 2; ++ni)
            bfrag[ni] = *(const v16bf*)(bbase + (size_t)ni * bstride
                                              + (size_t)kc * 512);
#pragma unroll
        for (int mi = 0; mi < 4; ++mi) {
            v16bf afrag =
                *(const v16bf*)(&ldsA[buf][((wm * 4 + mi) * 32 + lane) * 16]);
#pragma unroll
            for (int ni = 0; ni < 2; ++ni)
                acc[mi][ni] = __builtin_amdgcn_wmma_f32_16x16x32_bf16(
                    false, afrag, false, bfrag[ni], (short)0, acc[mi][ni],
                    false, false);
        }
        wait_async0();       // producer-side: back buffer filled before barrier
        __syncthreads();
    }

    // --- epilogue: C layout (VGPR v -> M = v + 8*(lane>=16), N = lane&15) ---
    int rsel = (lane & 16) ? 8 : 0;
    int coll = lane & 15;
#pragma unroll
    for (int mi = 0; mi < 4; ++mi) {
#pragma unroll
        for (int ni = 0; ni < 2; ++ni) {
            int colg = nbase + wn * 32 + ni * 16 + coll;
            float bv = bias[(size_t)e * Ndim + colg];
#pragma unroll
            for (int v = 0; v < 8; ++v) {
                int lr = wm * 64 + mi * 16 + rsel + v;
                if (mloc + lr < nrows) {
                    float val = acc[mi][ni][v] + bv;
                    if (FC1) {
                        // exact GELU: x*0.5*(1+erf(x/sqrt(2)))
                        val = 0.5f * val * (1.0f + erff(val * 0.70710678118f));
                        outb[(size_t)(gm + lr) * Ndim + colg] = f2bf(val);
                    } else {
                        outf[(size_t)(gm + lr) * Ndim + colg] = val;
                    }
                }
            }
        }
    }
}

// ---------------- combine: y[t] = w0*o2[row0] + w1*o2[row1] ----------------
__global__ __launch_bounds__(256) void combine_kernel(
    const float* __restrict__ o2, const int* __restrict__ row_of,
    const float* __restrict__ wrow, float* __restrict__ out) {
    const int qpd = DMODEL / 4;
    int i = blockIdx.x * blockDim.x + threadIdx.x;
    if (i >= TOKENS * qpd) return;
    int t = i / qpd, q = i % qpd;
    int r0 = row_of[t * 2 + 0], r1 = row_of[t * 2 + 1];
    float w0 = wrow[r0], w1 = wrow[r1];
    float4 a = ((const float4*)(o2 + (size_t)r0 * DMODEL))[q];
    float4 b = ((const float4*)(o2 + (size_t)r1 * DMODEL))[q];
    float4 y;
    y.x = w0 * a.x + w1 * b.x;
    y.y = w0 * a.y + w1 * b.y;
    y.z = w0 * a.z + w1 * b.z;
    y.w = w0 * a.w + w1 * b.w;
    ((float4*)(out + (size_t)t * DMODEL))[q] = y;
}

// ---------------- launcher ----------------
extern "C" void kernel_launch(void* const* d_in, const int* in_sizes, int n_in,
                              void* d_out, int out_size, void* d_ws,
                              size_t ws_size, hipStream_t stream) {
    (void)in_sizes; (void)n_in; (void)out_size; (void)ws_size;
    const float* x  = (const float*)d_in[0];
    const float* gw = (const float*)d_in[1];
    const float* w1 = (const float*)d_in[2];
    const float* b1 = (const float*)d_in[3];
    const float* w2 = (const float*)d_in[4];
    const float* b2 = (const float*)d_in[5];
    float* out = (float*)d_out;

    char* ws = (char*)d_ws;
    size_t off = 0;
    auto take = [&](size_t bytes) -> char* {
        char* p = ws + off;
        off += (bytes + 255) & ~(size_t)255;
        return p;
    };
    bf16r* xb   = (bf16r*)take((size_t)TOKENS * DMODEL * 2);
    bf16r* w1s  = (bf16r*)take((size_t)NEXP * DMODEL * HDIM * 2);
    bf16r* w2s  = (bf16r*)take((size_t)NEXP * HDIM * DMODEL * 2);
    bf16r* hbuf = (bf16r*)take((size_t)NROWS * HDIM * 2);
    float* o2   = (float*)take((size_t)NROWS * DMODEL * 4);
    int*   tidx = (int*)take((size_t)TOKENS * 2 * 4);
    float* tw   = (float*)take((size_t)TOKENS * 2 * 4);
    int* counts = (int*)take(64);
    int* offs   = (int*)take(64);
    int* curs   = (int*)take(64);
    int* token_of_row    = (int*)take((size_t)NROWS * 4);
    float* weight_of_row = (float*)take((size_t)NROWS * 4);
    int* row_of = (int*)take((size_t)TOKENS * 2 * 4);

    zero_counts_kernel<<<1, 32, 0, stream>>>(counts);
    cvt_bf16_kernel<<<2048, 256, 0, stream>>>(x, xb, TOKENS * DMODEL);
    swizzle_w_kernel<<<4096, 256, 0, stream>>>(w1, w1s, DMODEL, HDIM);
    swizzle_w_kernel<<<4096, 256, 0, stream>>>(w2, w2s, HDIM, DMODEL);
    gate_topk_kernel<<<TOKENS / 8, 256, 0, stream>>>(x, gw, tidx, tw, counts);
    scan_kernel<<<1, 32, 0, stream>>>(counts, offs, curs);
    assign_rows_kernel<<<TOKENS / 256, 256, 0, stream>>>(
        tidx, tw, curs, token_of_row, weight_of_row, row_of);
    moe_gemm_kernel<true><<<dim3(HDIM / 128, TOKENS / 128, NEXP), 256, 0,
                            stream>>>(xb, w1s, b1, hbuf, nullptr, offs,
                                      token_of_row, DMODEL, HDIM, DMODEL);
    moe_gemm_kernel<false><<<dim3(DMODEL / 128, TOKENS / 128, NEXP), 256, 0,
                             stream>>>(hbuf, w2s, b2, nullptr, o2, offs,
                                       nullptr, HDIM, DMODEL, HDIM);
    combine_kernel<<<(TOKENS * (DMODEL / 4)) / 256, 256, 0, stream>>>(
        o2, row_of, weight_of_row, out);
}
